// SagAttentionModule_40931038331120
// MI455X (gfx1250) — 16-bit and 8-bit matrix operand order repaired and checked by reading, compiled, not run on hardware
//
#include <hip/hip_runtime.h>
#include <stdint.h>


typedef __attribute__((ext_vector_type(2)))  int      vi2;
typedef __attribute__((ext_vector_type(4)))  int      vi4;
typedef __attribute__((ext_vector_type(4)))  float    vf4;
typedef __attribute__((ext_vector_type(8)))  int      v8i;
typedef __attribute__((ext_vector_type(8)))  float    v8f;
typedef __attribute__((ext_vector_type(4)))  _Float16 v4h;
typedef __attribute__((ext_vector_type(8)))  _Float16 v8h;
typedef __attribute__((ext_vector_type(16))) _Float16 v16h;

#define SEQ 2048
#define NH  16
#define HD  128
#define HID 2048
#define NB  2
static constexpr float SM_SCALE = 0.0883883476483184f;
#define MASKV (-3.0e38f)

#define Q8_OFF   ((size_t)0)
#define K8T_OFF  ((size_t)8388608)
#define V16_OFF  ((size_t)16777216)
#define QS_OFF   ((size_t)33554432)
#define KS_OFF   ((size_t)33562624)
#define KM_OFF   ((size_t)33566720)
#define VM_OFF   ((size_t)33583104)

__global__ void sage_means(const float* __restrict__ k, const float* __restrict__ v,
                           float* __restrict__ kmean, float* __restrict__ vmean) {
  int bh = blockIdx.x, b = bh >> 4, h = bh & 15;
  int t = threadIdx.x;
  const float* kp = k + (size_t)b * SEQ * HID + h * HD + t;
  const float* vp = v + (size_t)b * SEQ * HID + h * HD + t;
  float sk = 0.f, sv = 0.f;
  for (int s = 0; s < SEQ; ++s) {
    sk += kp[(size_t)s * HID];
    sv += vp[(size_t)s * HID];
  }
  kmean[bh * HD + t] = sk * (1.f / SEQ);
  vmean[bh * HD + t] = sv * (1.f / SEQ);
}

__global__ void sage_quant_q(const float* __restrict__ q, int8_t* __restrict__ q8,
                             float* __restrict__ qscale) {
  __shared__ float red[256];
  int blk = blockIdx.x, bh = blk >> 6, g = blk & 63;
  int b = bh >> 4, h = bh & 15;
  int tid = threadIdx.x;
  float vals[16];
  float am = 0.f;
#pragma unroll
  for (int i = 0; i < 16; ++i) {
    int e = i * 256 + tid;
    int row = e >> 7, dim = e & 127;
    float x = q[((size_t)(b * SEQ + g * 32 + row)) * HID + h * HD + dim];
    vals[i] = x;
    am = fmaxf(am, fabsf(x));
  }
  red[tid] = am; __syncthreads();
  for (int s = 128; s > 0; s >>= 1) {
    if (tid < s) red[tid] = fmaxf(red[tid], red[tid + s]);
    __syncthreads();
  }
  float scale = fmaxf(red[0] * (1.f / 127.f), 1e-8f);
  if (tid == 0) qscale[bh * 64 + g] = scale;
  float inv = 1.f / scale;
#pragma unroll
  for (int i = 0; i < 16; ++i) {
    int e = i * 256 + tid;
    int row = e >> 7, dim = e & 127;
    float r = fminf(fmaxf(rintf(vals[i] * inv), -127.f), 127.f);
    q8[((size_t)bh * SEQ + g * 32 + row) * HD + dim] = (int8_t)r;
  }
}

__global__ void sage_quant_k(const float* __restrict__ k, const float* __restrict__ kmean,
                             int8_t* __restrict__ k8t, float* __restrict__ kscale) {
  __shared__ float red[256];
  int blk = blockIdx.x, bh = blk >> 5, g = blk & 31;
  int b = bh >> 4, h = bh & 15;
  int tid = threadIdx.x;
  float vals[32];
  float am = 0.f;
#pragma unroll
  for (int i = 0; i < 32; ++i) {
    int e = i * 256 + tid;
    int row = e >> 7, dim = e & 127;
    float x = k[((size_t)(b * SEQ + g * 64 + row)) * HID + h * HD + dim]
              - kmean[bh * HD + dim];
    vals[i] = x;
    am = fmaxf(am, fabsf(x));
  }
  red[tid] = am; __syncthreads();
  for (int s = 128; s > 0; s >>= 1) {
    if (tid < s) red[tid] = fmaxf(red[tid], red[tid + s]);
    __syncthreads();
  }
  float scale = fmaxf(red[0] * (1.f / 127.f), 1e-8f);
  if (tid == 0) kscale[bh * 32 + g] = scale;
  float inv = 1.f / scale;
#pragma unroll
  for (int i = 0; i < 32; ++i) {
    int e = i * 256 + tid;
    int row = e >> 7, dim = e & 127;
    float r = fminf(fmaxf(rintf(vals[i] * inv), -127.f), 127.f);
    k8t[((size_t)bh * HD + dim) * SEQ + g * 64 + row] = (int8_t)r;
  }
}

__global__ void sage_smooth_v(const float* __restrict__ v, const float* __restrict__ vmean,
                              _Float16* __restrict__ v16) {
  size_t e = ((size_t)blockIdx.x * 256 + threadIdx.x) * 4;
  int bh = (int)(e >> 18);
  size_t rem = e & (size_t)262143;
  int row = (int)(rem >> 7), dim = (int)(rem & 127);
  int b = bh >> 4, h = bh & 15;
  vf4 x = *(const vf4*)(v + ((size_t)(b * SEQ + row)) * HID + h * HD + dim);
  const float* vm = vmean + bh * HD + dim;
  v4h o;
  o[0] = (_Float16)(x[0] - vm[0]);
  o[1] = (_Float16)(x[1] - vm[1]);
  o[2] = (_Float16)(x[2] - vm[2]);
  o[3] = (_Float16)(x[3] - vm[3]);
  *(v4h*)(v16 + (size_t)bh * (SEQ * HD) + (size_t)row * HD + dim) = o;
}

__global__ __launch_bounds__(256) void sage_attn(
    const int8_t* __restrict__ q8, const int8_t* __restrict__ k8t,
    const _Float16* __restrict__ v16, const float* __restrict__ qscale,
    const float* __restrict__ kscale, const float* __restrict__ vmean,
    const int* __restrict__ wleft, const int* __restrict__ wright,
    float* __restrict__ out) {
  __shared__ _Float16 pbuf[8][16 * 32];

  const int wl = wleft[0];
  const int wr = wright[0];

  int bh   = blockIdx.x >> 4;
  int tile = blockIdx.x & 15;
  int b = bh >> 4, h = bh & 15;
  int wid  = threadIdx.x >> 5;
  int lane = threadIdx.x & 31;
  int m  = lane & 15;
  int hi = lane >> 4;
  int q0 = tile * 128 + wid * 16;

  const int8_t* qrow = q8 + ((size_t)bh * SEQ + q0 + m) * HD + hi * 8;
  v8i aq[2];
#pragma unroll
  for (int half = 0; half < 2; ++half) {
    const int8_t* p = qrow + half * 64;
    vi2 p0 = *(const vi2*)(p + 0);
    vi2 p1 = *(const vi2*)(p + 16);
    vi2 p2 = *(const vi2*)(p + 32);
    vi2 p3 = *(const vi2*)(p + 48);
    v8i a;
    a[0] = p0[0]; a[1] = p0[1]; a[2] = p1[0]; a[3] = p1[1];
    a[4] = p2[0]; a[5] = p2[1]; a[6] = p3[0]; a[7] = p3[1];
    aq[half] = a;
  }
  float qs = qscale[bh * 64 + (q0 >> 5)];

  const int8_t*   ktb = k8t + (size_t)bh * HD * SEQ;
  const _Float16* vb  = v16 + (size_t)bh * SEQ * HD;

  float runm[8], runl[8];
  v8f acc[8];
  v8f zf = {};
#pragma unroll
  for (int r = 0; r < 8; ++r) { runm[r] = MASKV; runl[r] = 0.f; }
#pragma unroll
  for (int t = 0; t < 8; ++t) acc[t] = zf;

  int jlo = 0, jhi = SEQ;
  if (wl >= 0) { jlo = q0 - wl; if (jlo < 0) jlo = 0; jlo &= ~31; }
  if (wr >= 0) { jhi = q0 + 16 + wr; jhi = (jhi + 31) & ~31; if (jhi > SEQ) jhi = SEQ; }

  for (int j0 = jlo; j0 < jhi; j0 += 32) {
    float ksc = kscale[bh * 32 + (j0 >> 6)];
    float dq  = qs * ksc * SM_SCALE;

    v8i sacc[2];
#pragma unroll
    for (int sub = 0; sub < 2; ++sub) {
      int jb = j0 + sub * 16;
      v8i c = {};
#pragma unroll
      for (int half = 0; half < 2; ++half) {
        const int8_t* kp = ktb + ((size_t)(half * 64 + hi * 8)) * SEQ + jb + m;
        v8i bk;
#pragma unroll
        for (int i = 0; i < 8; ++i) {
          unsigned w = 0;
#pragma unroll
          for (int bb = 0; bb < 4; ++bb) {
            int s = i * 4 + bb;
            int kk = (s >> 3) * 16 + (s & 7);
            w |= (unsigned)(uint8_t)kp[(size_t)kk * SEQ] << (8 * bb);
          }
          bk[i] = (int)w;
        }
        c = __builtin_amdgcn_wmma_i32_16x16x64_iu8(true, aq[half], true, bk, c,
                                                   false, false);
      }
      sacc[sub] = c;
    }

    float f[2][8];
#pragma unroll
    for (int sub = 0; sub < 2; ++sub) {
      int j = j0 + sub * 16 + m;
#pragma unroll
      for (int r = 0; r < 8; ++r) {
        int row = q0 + r + hi * 8;
        bool ok = (wl < 0 || (row - j) <= wl) && (wr < 0 || (j - row) <= wr);
        f[sub][r] = ok ? (float)sacc[sub][r] * dq : MASKV;
      }
    }

#pragma unroll
    for (int r = 0; r < 8; ++r) {
      float bm = fmaxf(f[0][r], f[1][r]);
      bm = fmaxf(bm, __shfl_xor(bm, 1, 32));
      bm = fmaxf(bm, __shfl_xor(bm, 2, 32));
      bm = fmaxf(bm, __shfl_xor(bm, 4, 32));
      bm = fmaxf(bm, __shfl_xor(bm, 8, 32));
      float nm = fmaxf(runm[r], bm);
      float sc = __expf(runm[r] - nm);
      runm[r] = nm;
      float p0 = __expf(f[0][r] - nm);
      float p1 = __expf(f[1][r] - nm);
      f[0][r] = p0; f[1][r] = p1;
      float rs = p0 + p1;
      rs += __shfl_xor(rs, 1, 32);
      rs += __shfl_xor(rs, 2, 32);
      rs += __shfl_xor(rs, 4, 32);
      rs += __shfl_xor(rs, 8, 32);
      runl[r] = runl[r] * sc + rs;
#pragma unroll
      for (int t = 0; t < 8; ++t) acc[t][r] *= sc;
    }

    _Float16* pl = pbuf[wid];
#pragma unroll
    for (int r = 0; r < 8; ++r) {
      int row = r + hi * 8;
      pl[row * 32 + m]      = (_Float16)f[0][r];
      pl[row * 32 + 16 + m] = (_Float16)f[1][r];
    }
    asm volatile("s_wait_dscnt 0" ::: "memory");

    const _Float16* pr = pl + m * 32 + hi * 8;
    v8h plo = *(const v8h*)pr;
    v8h phi = *(const v8h*)(pr + 16);
    v16h pa = __builtin_shufflevector(plo, phi, 0, 1, 2, 3, 4, 5, 6, 7,
                                      8, 9, 10, 11, 12, 13, 14, 15);

    const _Float16* vrow = vb + (size_t)(j0 + hi * 8) * HD + m;
#pragma unroll
    for (int t = 0; t < 8; ++t) {
      v16h bv;
#pragma unroll
      for (int s = 0; s < 8; ++s) {
        bv[s]     = vrow[(size_t)s * HD + t * 16];
        bv[8 + s] = vrow[(size_t)(16 + s) * HD + t * 16];
      }
      acc[t] = __builtin_amdgcn_wmma_f32_16x16x32_f16(false, pa, false, bv,
                                                      (short)0, acc[t],
                                                      false, false);
    }
  }

  float inv[8];
#pragma unroll
  for (int r = 0; r < 8; ++r) inv[r] = 1.f / runl[r];
#pragma unroll
  for (int t = 0; t < 8; ++t) {
    int dim = t * 16 + m;
    float vm = vmean[bh * HD + dim];
#pragma unroll
    for (int r = 0; r < 8; ++r) {
      int row = q0 + r + hi * 8;
      out[((size_t)(b * SEQ + row)) * HID + h * HD + dim] = acc[t][r] * inv[r] + vm;
    }
  }
}

extern "C" void kernel_launch(void* const* d_in, const int* in_sizes, int n_in,
                              void* d_out, int out_size, void* d_ws, size_t ws_size,
                              hipStream_t stream) {
  (void)in_sizes; (void)n_in; (void)out_size; (void)ws_size;
  const float* q = (const float*)d_in[0];
  const float* k = (const float*)d_in[1];
  const float* v = (const float*)d_in[2];
  const int* wl = (const int*)d_in[3];
  const int* wr = (const int*)d_in[4];
  float* out = (float*)d_out;

  char* ws = (char*)d_ws;
  int8_t*   q8  = (int8_t*)(ws + Q8_OFF);
  int8_t*   k8t = (int8_t*)(ws + K8T_OFF);
  _Float16* v16 = (_Float16*)(ws + V16_OFF);
  float* qs = (float*)(ws + QS_OFF);
  float* ks = (float*)(ws + KS_OFF);
  float* km = (float*)(ws + KM_OFF);
  float* vm = (float*)(ws + VM_OFF);

  sage_means  <<<NB * NH,        128, 0, stream>>>(k, v, km, vm);
  sage_quant_q<<<NB * NH * 64,   256, 0, stream>>>(q, q8, qs);
  sage_quant_k<<<NB * NH * 32,   256, 0, stream>>>(k, km, k8t, ks);
  sage_smooth_v<<<8192,          256, 0, stream>>>(v, vm, v16);
  sage_attn   <<<NB * NH * 16,   256, 0, stream>>>(q8, k8t, v16, qs, ks, vm, wl, wr, out);
}
